// CAUSETRModel_7490422964310
// MI455X (gfx1250) — matrix operand order repaired and checked by reading, not compiled, not run on hardware
//
#include <hip/hip_runtime.h>
#include <hip/hip_bf16.h>
#include <math.h>

// ---------------------------------------------------------------------------
// CDNA5 (gfx1250) wave32 WMMA bf16 GEMM-based implementation.
// Block tile 128x128 (8 waves, 32x64 per wave, 8 WMMA per K-step),
// double-buffered LDS, async global->LDS (ASYNCcnt) for NT tiles.
// ---------------------------------------------------------------------------

#if __has_builtin(__builtin_amdgcn_global_load_async_to_lds_b128) && \
    __has_builtin(__builtin_amdgcn_s_wait_asynccnt)
#define HAVE_ASYNC 1
#endif

#define AS1 __attribute__((address_space(1)))
#define AS3 __attribute__((address_space(3)))

typedef int v4i __attribute__((vector_size(16)));   // matches builtin param type

typedef __attribute__((ext_vector_type(16))) __bf16 v16bf;
typedef __attribute__((ext_vector_type(8)))  float  v8f;

union Frag16 { v16bf v; uint4 q[2]; };

#define BMT 128   // block tile M
#define BNT 128   // block tile N
#define BKT 32    // K step (one bf16 WMMA K)
#define LDP 40    // LDS row stride in bf16 elems (80B, 16B-aligned, skewed)

struct GemmP {
  const __bf16* A; long long lda, sA;          // A[M,K] row-major (elems)
  const __bf16* B; long long ldb, sB;          // BTRANS=1: B[N,K]; 0: B[K,N]
  int nSrc;                                    // valid B rows(n); beyond -> clamped
  const float* bias;                           // len >= nLimit or null
  float alpha; int relu;
  float* Cf;  long long ldc,  sC;              // optional f32 out
  __bf16* Cb; long long ldcb, sCb;             // optional bf16 out
  float* Cperm; int permN;                     // optional (s,b)->(b,s) permuted f32 out
  int M, N, K, nLimit;
};

template <int BTRANS>
__global__ __launch_bounds__(256) void gemm_bf16_wmma(GemmP p) {
  __shared__ __align__(16) __bf16 As[2][BMT * LDP];
  __shared__ __align__(16) __bf16 Bs[2][BNT * LDP];

  const int tid = threadIdx.x;
  const int z   = blockIdx.z;
  const __bf16* A  = p.A + (long long)z * p.sA;
  const __bf16* Bm = p.B + (long long)z * p.sB;
  float*  Cf = p.Cf ? p.Cf + (long long)z * p.sC  : (float*)0;
  __bf16* Cb = p.Cb ? p.Cb + (long long)z * p.sCb : (__bf16*)0;

  const int m0 = blockIdx.y * BMT;
  const int n0 = blockIdx.x * BNT;

  const int w    = tid >> 5;        // wave 0..7
  const int lane = tid & 31;
  const int wm   = w >> 1;          // 0..3  (32-row strip)
  const int wn   = w & 1;           // 0..1  (64-col strip)
  const int lm   = lane & 15;
  const int hi   = lane >> 4;

  // per-thread 16B tile-piece coordinates.
  // A/B(NT): piece idx = it*256+tid -> row = 64*it + tid>>2, kg = (tid&3)*8.
  const int pRow = tid >> 2;
  const int pKg  = (tid & 3) * 8;
  const int aM0  = (m0 + pRow      < p.M) ? m0 + pRow      : p.M - 1;   // clamp: OOB rows
  const int aM1  = (m0 + 64 + pRow < p.M) ? m0 + 64 + pRow : p.M - 1;   // only feed unstored C
  const int bN0  = (n0 + pRow      < p.nSrc) ? n0 + pRow      : p.nSrc - 1;
  const int bN1  = (n0 + 64 + pRow < p.nSrc) ? n0 + 64 + pRow : p.nSrc - 1;
  // B(NN): piece idx = it*256+tid -> k = 16*it + tid>>4, ng = (tid&15)*8.
  const int kNN0 = tid >> 4;
  const int kNN1 = 16 + (tid >> 4);
  const int ngNN = (tid & 15) * 8;

  v8f acc[2][4] = {};
  const int T = p.K / BKT;

  // ---- fragment build + 8 WMMAs from LDS buffer b ----
  auto computeTile = [&](int b) {
    Frag16 af[2], bf4[4];
#pragma unroll
    for (int mi = 0; mi < 2; ++mi) {
      const __bf16* bp = &As[b][(wm * 32 + mi * 16 + lm) * LDP];
      af[mi].q[0] = *(const uint4*)(bp + hi * 8);       // elems 0..7  : K = 8*hi + i
      af[mi].q[1] = *(const uint4*)(bp + 16 + hi * 8);  // elems 8..15 : K = 16 + 8*hi + i
    }
#pragma unroll
    for (int ni = 0; ni < 4; ++ni) {
      const __bf16* bp = &Bs[b][(wn * 64 + ni * 16 + lm) * LDP];
      bf4[ni].q[0] = *(const uint4*)(bp + hi * 8);      // elems 0..7  : K = 8*hi + i
      bf4[ni].q[1] = *(const uint4*)(bp + 16 + hi * 8); // elems 8..15 : K = 16 + 8*hi + i
    }
#pragma unroll
    for (int mi = 0; mi < 2; ++mi)
#pragma unroll
      for (int ni = 0; ni < 4; ++ni)
        acc[mi][ni] = __builtin_amdgcn_wmma_f32_16x16x32_bf16(
            false, af[mi].v, false, bf4[ni].v, (short)0, acc[mi][ni], false, false);
  };

#if defined(HAVE_ASYNC)
  auto awaitAsync = [&]() { __builtin_amdgcn_s_wait_asynccnt(0); };
  auto issueAsync = [&](int t, int b) {
    const long long k0 = (long long)t * BKT;
    __builtin_amdgcn_global_load_async_to_lds_b128(
        (AS1 v4i*)(A + (long long)aM0 * p.lda + k0 + pKg),
        (AS3 v4i*)&As[b][pRow * LDP + pKg], 0, 0);
    __builtin_amdgcn_global_load_async_to_lds_b128(
        (AS1 v4i*)(A + (long long)aM1 * p.lda + k0 + pKg),
        (AS3 v4i*)&As[b][(64 + pRow) * LDP + pKg], 0, 0);
    __builtin_amdgcn_global_load_async_to_lds_b128(
        (AS1 v4i*)(Bm + (long long)bN0 * p.ldb + k0 + pKg),
        (AS3 v4i*)&Bs[b][pRow * LDP + pKg], 0, 0);
    __builtin_amdgcn_global_load_async_to_lds_b128(
        (AS1 v4i*)(Bm + (long long)bN1 * p.ldb + k0 + pKg),
        (AS3 v4i*)&Bs[b][(64 + pRow) * LDP + pKg], 0, 0);
  };
#endif

  // register-staged path (NN always; NT when async builtins are unavailable)
  uint4 rA0, rA1, rB0, rB1;
  auto fetch = [&](int t) {
    const long long k0 = (long long)t * BKT;
    rA0 = *(const uint4*)(A + (long long)aM0 * p.lda + k0 + pKg);
    rA1 = *(const uint4*)(A + (long long)aM1 * p.lda + k0 + pKg);
    if (BTRANS) {
      rB0 = *(const uint4*)(Bm + (long long)bN0 * p.ldb + k0 + pKg);
      rB1 = *(const uint4*)(Bm + (long long)bN1 * p.ldb + k0 + pKg);
    } else {
      rB0 = *(const uint4*)(Bm + (k0 + kNN0) * p.ldb + n0 + ngNN);
      rB1 = *(const uint4*)(Bm + (k0 + kNN1) * p.ldb + n0 + ngNN);
    }
    if (t + 1 < T)  // next-slab prefetch (global_prefetch_b8)
      __builtin_prefetch(A + (long long)aM0 * p.lda + k0 + BKT + pKg, 0, 3);
  };
  auto commit = [&](int b) {
    *(uint4*)&As[b][pRow * LDP + pKg]        = rA0;
    *(uint4*)&As[b][(64 + pRow) * LDP + pKg] = rA1;
    if (BTRANS) {
      *(uint4*)&Bs[b][pRow * LDP + pKg]        = rB0;
      *(uint4*)&Bs[b][(64 + pRow) * LDP + pKg] = rB1;
    } else {
      union { uint4 q; __bf16 e[8]; } u0, u1;
      u0.q = rB0; u1.q = rB1;
#pragma unroll
      for (int j = 0; j < 8; ++j) Bs[b][(ngNN + j) * LDP + kNN0] = u0.e[j];
#pragma unroll
      for (int j = 0; j < 8; ++j) Bs[b][(ngNN + j) * LDP + kNN1] = u1.e[j];
    }
  };

#if defined(HAVE_ASYNC)
  if (BTRANS == 1) {
    issueAsync(0, 0);
    awaitAsync();
    __syncthreads();
    for (int t = 0; t < T; ++t) {
      if (t + 1 < T) issueAsync(t + 1, (t + 1) & 1);   // overlap with WMMA below
      computeTile(t & 1);
      if (t + 1 < T) awaitAsync();
      __syncthreads();
    }
  } else
#endif
  {
    fetch(0);
    commit(0);
    __syncthreads();
    for (int t = 0; t < T; ++t) {
      if (t + 1 < T) fetch(t + 1);                     // global loads in flight
      computeTile(t & 1);
      if (t + 1 < T) commit((t + 1) & 1);
      __syncthreads();
    }
  }

  // ---- epilogue: C/D layout -> M = j + 8*hi, N = lane&15 ----
#pragma unroll
  for (int mi = 0; mi < 2; ++mi) {
#pragma unroll
    for (int ni = 0; ni < 4; ++ni) {
      int n = n0 + wn * 64 + ni * 16 + lm;
      if (n >= p.nLimit) continue;
      float bv = p.bias ? p.bias[n] : 0.0f;
#pragma unroll
      for (int j = 0; j < 8; ++j) {
        int m = m0 + wm * 32 + mi * 16 + hi * 8 + j;
        if (m >= p.M) continue;
        float v = acc[mi][ni][j] * p.alpha + bv;
        if (p.relu && v < 0.0f) v = 0.0f;
        if (Cf)       Cf[(long long)m * p.ldc  + n] = v;
        if (Cb)       Cb[(long long)m * p.ldcb + n] = (__bf16)v;
        if (p.Cperm)  p.Cperm[(((long long)(m & 15)) * 784 + (m >> 4)) * p.permN + n] = v;
      }
    }
  }
}

// ---------------------------------------------------------------------------
// Support kernels
// ---------------------------------------------------------------------------

__global__ __launch_bounds__(256) void cvt_pad_bf16(const float* s, __bf16* d,
                                                    long long rows, int cs, int cd) {
  long long n = rows * (long long)cd;
  for (long long i = (long long)blockIdx.x * 256 + threadIdx.x; i < n;
       i += (long long)gridDim.x * 256) {
    long long r = i / cd;
    int c = (int)(i - r * cd);
    d[i] = (c < cs) ? (__bf16)s[r * cs + c] : (__bf16)0.0f;
  }
}

__global__ __launch_bounds__(256) void zero_bf16_k(__bf16* p, long long n) {
  for (long long i = (long long)blockIdx.x * 256 + threadIdx.x; i < n;
       i += (long long)gridDim.x * 256) p[i] = (__bf16)0.0f;
}

// feat (B,P,D) -> mem bf16 in (s*16+b, D) layout
__global__ __launch_bounds__(256) void feat_to_mem(const float* feat, __bf16* memb,
                                                   long long MD) {
  for (long long i = (long long)blockIdx.x * 256 + threadIdx.x; i < MD;
       i += (long long)gridDim.x * 256) {
    int dd = (int)(i & 1023);
    long long r = i >> 10;
    int s = (int)(r >> 4), b = (int)(r & 15);
    memb[i] = (__bf16)feat[(((long long)b * 784) + s) * 1024 + dd];
  }
}

// one block per row: y = bf16(x / ||x||2)
__global__ __launch_bounds__(256) void rownorm_bf16(const float* x, __bf16* y, int cols) {
  __shared__ float red[256];
  long long r = blockIdx.x;
  int tid = threadIdx.x;
  float ss = 0.0f;
  for (int c = tid; c < cols; c += 256) { float v = x[r * cols + c]; ss += v * v; }
  red[tid] = ss; __syncthreads();
  for (int s = 128; s > 0; s >>= 1) { if (tid < s) red[tid] += red[tid + s]; __syncthreads(); }
  float inv = rsqrtf(red[0]);
  __syncthreads();
  for (int c = tid; c < cols; c += 256) y[r * cols + c] = (__bf16)(x[r * cols + c] * inv);
}

__global__ __launch_bounds__(256) void argmax_row(const float* sc, int cols, int* out) {
  __shared__ float rv[256];
  __shared__ int   ri[256];
  long long r = blockIdx.x;
  int tid = threadIdx.x;
  float best = -1e30f; int bi = 0;
  for (int c = tid; c < cols; c += 256) {
    float v = sc[r * cols + c];
    if (v > best) { best = v; bi = c; }
  }
  rv[tid] = best; ri[tid] = bi; __syncthreads();
  for (int s = 128; s > 0; s >>= 1) {
    if (tid < s && rv[tid + s] > rv[tid]) { rv[tid] = rv[tid + s]; ri[tid] = ri[tid + s]; }
    __syncthreads();
  }
  if (tid == 0) out[r] = ri[0];
}

// tgt = codebook[idx] in (s*16+b) layout; also bf16 tgt and bf16 (tgt+pos)
__global__ __launch_bounds__(256) void gather_pos(const float* cb, const int* idx,
                                                  const float* pos, float* res,
                                                  __bf16* curb, __bf16* qkb) {
  int r = blockIdx.x;
  int b = r & 15, s = r >> 4;
  int c = idx[(long long)b * 784 + s];
  for (int d = threadIdx.x; d < 1024; d += 256) {
    float v = cb[(long long)c * 1024 + d];
    res[(long long)r * 1024 + d]  = v;
    curb[(long long)r * 1024 + d] = (__bf16)v;
    qkb[(long long)r * 1024 + d]  = (__bf16)(v + pos[(long long)s * 1024 + d]);
  }
}

// softmax over T, write bf16 padded to Tp (zeros in [T,Tp))
__global__ __launch_bounds__(256) void softmax_row(const float* sc, __bf16* attn,
                                                   int T, int Tp) {
  __shared__ float red[256];
  long long row = blockIdx.x;
  int tid = threadIdx.x;
  const float* x = sc + row * T;
  float mx = -1e30f;
  for (int t = tid; t < T; t += 256) mx = fmaxf(mx, x[t]);
  red[tid] = mx; __syncthreads();
  for (int s = 128; s > 0; s >>= 1) { if (tid < s) red[tid] = fmaxf(red[tid], red[tid + s]); __syncthreads(); }
  mx = red[0]; __syncthreads();
  float sum = 0.0f;
  for (int t = tid; t < T; t += 256) sum += __expf(x[t] - mx);
  red[tid] = sum; __syncthreads();
  for (int s = 128; s > 0; s >>= 1) { if (tid < s) red[tid] += red[tid + s]; __syncthreads(); }
  float inv = 1.0f / red[0];
  for (int t = tid; t < Tp; t += 256)
    attn[row * Tp + t] = (t < T) ? (__bf16)(__expf(x[t] - mx) * inv) : (__bf16)0.0f;
}

// res = LN(res + add)*g + b  (+ optional transposed feat add); emit bf16 copies
__global__ __launch_bounds__(256) void ln_res(float* res, const float* add,
                                              const float* g, const float* beta,
                                              __bf16* curb, __bf16* qkb,
                                              const float* pos, const float* memAdd) {
  __shared__ float red[256];
  int r = blockIdx.x, tid = threadIdx.x;
  float xs[4];
  float s1 = 0.0f;
#pragma unroll
  for (int i = 0; i < 4; ++i) {
    int d = tid + i * 256;
    float x = res[(long long)r * 1024 + d] + add[(long long)r * 1024 + d];
    xs[i] = x; s1 += x;
  }
  red[tid] = s1; __syncthreads();
  for (int s = 128; s > 0; s >>= 1) { if (tid < s) red[tid] += red[tid + s]; __syncthreads(); }
  float mu = red[0] * (1.0f / 1024.0f);
  __syncthreads();
  float s2 = 0.0f;
#pragma unroll
  for (int i = 0; i < 4; ++i) { float dv = xs[i] - mu; s2 += dv * dv; }
  red[tid] = s2; __syncthreads();
  for (int s = 128; s > 0; s >>= 1) { if (tid < s) red[tid] += red[tid + s]; __syncthreads(); }
  float rstd = rsqrtf(red[0] * (1.0f / 1024.0f) + 1e-5f);
  __syncthreads();
  int b = r & 15, sIdx = r >> 4;
#pragma unroll
  for (int i = 0; i < 4; ++i) {
    int d = tid + i * 256;
    float y = g[d] * (xs[i] - mu) * rstd + beta[d];
    if (memAdd) y += memAdd[(((long long)b * 784) + sIdx) * 1024 + d];
    res[(long long)r * 1024 + d]  = y;
    curb[(long long)r * 1024 + d] = (__bf16)y;
    if (qkb) qkb[(long long)r * 1024 + d] = (__bf16)(y + pos[(long long)sIdx * 1024 + d]);
  }
}

// tr = b1 + b2 (n<90); store permuted f32 out and bf16 padded to 96
__global__ __launch_bounds__(128) void combine_tr(const float* b1, const float* b2,
                                                  float* out0, __bf16* trb) {
  int r = blockIdx.x;
  int n = threadIdx.x;
  if (n >= 96) return;
  float v = 0.0f;
  if (n < 90) v = b1[(long long)r * 96 + n] + b2[(long long)r * 96 + n];
  trb[(long long)r * 96 + n] = (__bf16)v;
  if (n < 90) {
    int b = r & 15, s = r >> 4;
    out0[(((long long)b * 784) + s) * 90 + n] = v;
  }
}

// ---------------------------------------------------------------------------
// Host orchestration
// ---------------------------------------------------------------------------
extern "C" void kernel_launch(void* const* d_in, const int* in_sizes, int n_in,
                              void* d_out, int out_size, void* d_ws, size_t ws_size,
                              hipStream_t stream) {
  (void)in_sizes; (void)n_in; (void)out_size;
  const int BATCH = 16, S = 784, D = 1024, NC = 2048, HID = 2048, RD = 90, RDP = 96, PD = 2048;
  const int M = BATCH * S;                 // 12544
  const int SP = 800;                      // padded T (multiple of 32)
  const long long MD = (long long)M * D;

  const float* feat      = (const float*)d_in[0];
  const float* codebook  = (const float*)d_in[1];
  const float* query_pos = (const float*)d_in[2];
  const float* sa_wi     = (const float*)d_in[3];
  const float* sa_bi     = (const float*)d_in[4];
  const float* sa_wo     = (const float*)d_in[5];
  const float* sa_bo     = (const float*)d_in[6];
  const float* ca_wi     = (const float*)d_in[7];
  const float* ca_bi     = (const float*)d_in[8];
  const float* ca_wo     = (const float*)d_in[9];
  const float* ca_bo     = (const float*)d_in[10];
  const float* lin1_w    = (const float*)d_in[11];
  const float* lin1_bias = (const float*)d_in[12];
  const float* lin2_w    = (const float*)d_in[13];
  const float* lin2_bias = (const float*)d_in[14];
  const float* n1_g = (const float*)d_in[15];
  const float* n1_b = (const float*)d_in[16];
  const float* n2_g = (const float*)d_in[17];
  const float* n2_b = (const float*)d_in[18];
  const float* n3_g = (const float*)d_in[19];
  const float* n3_b = (const float*)d_in[20];
  const float* f1_w  = (const float*)d_in[21];
  const float* f1_b  = (const float*)d_in[22];
  const float* f2a_w = (const float*)d_in[23];
  const float* f2a_b = (const float*)d_in[24];
  const float* f2b_w = (const float*)d_in[25];
  const float* f2b_b = (const float*)d_in[26];
  const float* ph_w  = (const float*)d_in[27];
  const float* ph_b  = (const float*)d_in[28];

  float* out_tr   = (float*)d_out;                                // (B,P,90)
  float* out_proj = out_tr + (long long)BATCH * S * RD;           // (B,P,2048)

  // ---- workspace arena ----
  char* ws = (char*)d_ws;
  size_t cur = 0;
  auto alloc = [&](size_t bytes) -> char* {
    size_t a = (cur + 255) & ~(size_t)255;
    cur = a + bytes;
    return ws + a;
  };
  const size_t MDbf = (size_t)MD * 2, MDf = (size_t)MD * 4;
  __bf16* memb  = (__bf16*)alloc(MDbf);
  float*  res   = (float*) alloc(MDf);
  __bf16* curb  = (__bf16*)alloc(MDbf);
  __bf16* qkb   = (__bf16*)alloc(MDbf);
  int*    idx   = (int*)   alloc((size_t)M * 4);
  __bf16* sawiB = (__bf16*)alloc((size_t)3 * D * D * 2);
  __bf16* sawoB = (__bf16*)alloc((size_t)D * D * 2);
  __bf16* cawiB = (__bf16*)alloc((size_t)3 * D * D * 2);
  __bf16* cawoB = (__bf16*)alloc((size_t)D * D * 2);
  __bf16* l1B   = (__bf16*)alloc((size_t)HID * D * 2);
  __bf16* l2B   = (__bf16*)alloc((size_t)D * HID * 2);
  __bf16* f1B   = (__bf16*)alloc((size_t)RD * D * 2);
  __bf16* f2aB  = (__bf16*)alloc((size_t)D * D * 2);
  __bf16* f2bB  = (__bf16*)alloc((size_t)RD * D * 2);
  __bf16* phB   = (__bf16*)alloc((size_t)PD * RDP * 2);
  const size_t ARENA = (size_t)150 * 1024 * 1024;
  char* arena = alloc(ARENA);
  if (cur > ws_size) return;  // insufficient workspace (compile-only environment)

  dim3 t256(256);

  auto cvt = [&](const float* s, __bf16* d, long long rows, int cs, int cd) {
    cvt_pad_bf16<<<dim3(1024), t256, 0, stream>>>(s, d, rows, cs, cd);
  };

  auto G = [&](const __bf16* A, long long lda, long long sA,
               const __bf16* Bp, long long ldb, long long sB, int bTrans, int nSrc,
               const float* bias, float alpha, int relu,
               float* Cf, long long ldc, long long sC,
               __bf16* Cb, long long ldcb, long long sCb,
               float* Cperm, int permN,
               int Mm, int Nn, int Kk, int nLimit, int batches) {
    GemmP p;
    p.A = A; p.lda = lda; p.sA = sA;
    p.B = Bp; p.ldb = ldb; p.sB = sB; p.nSrc = nSrc;
    p.bias = bias; p.alpha = alpha; p.relu = relu;
    p.Cf = Cf; p.ldc = ldc; p.sC = sC;
    p.Cb = Cb; p.ldcb = ldcb; p.sCb = sCb;
    p.Cperm = Cperm; p.permN = permN;
    p.M = Mm; p.N = Nn; p.K = Kk; p.nLimit = nLimit;
    dim3 g((Nn + BNT - 1) / BNT, (Mm + BMT - 1) / BMT, batches);
    if (bTrans) gemm_bf16_wmma<1><<<g, t256, 0, stream>>>(p);
    else        gemm_bf16_wmma<0><<<g, t256, 0, stream>>>(p);
  };

  // ---- phase 0: weight conversions ----
  cvt(sa_wi, sawiB, (long long)3 * D, D, D);
  cvt(sa_wo, sawoB, D, D, D);
  cvt(ca_wi, cawiB, (long long)3 * D, D, D);
  cvt(ca_wo, cawoB, D, D, D);
  cvt(lin1_w, l1B, HID, D, D);
  cvt(lin2_w, l2B, D, HID, HID);
  cvt(f1_w,  f1B,  RD, D, D);
  cvt(f2a_w, f2aB, D, D, D);
  cvt(f2b_w, f2bB, RD, D, D);
  cvt(ph_w,  phB,  PD, RD, RDP);                       // zero-padded K 90->96
  feat_to_mem<<<dim3(2048), t256, 0, stream>>>(feat, memb, MD);

  // ---- phase 1: VQ (rows in feat order r' = b*784+s) ----
  {
    __bf16* nzb  = (__bf16*)(arena);
    __bf16* ncb  = (__bf16*)(arena + MDbf);
    float*  vqsc = (float*)(arena + MDbf + (size_t)NC * D * 2);
    rownorm_bf16<<<dim3(M),  t256, 0, stream>>>(feat, nzb, D);
    rownorm_bf16<<<dim3(NC), t256, 0, stream>>>(codebook, ncb, D);
    G(nzb, D, 0, ncb, D, 0, 1, NC, nullptr, 1.0f, 0,
      vqsc, NC, 0, nullptr, 0, 0, nullptr, 0, M, NC, D, NC, 1);
    argmax_row<<<dim3(M), t256, 0, stream>>>(vqsc, NC, idx);
  }
  gather_pos<<<dim3(M), t256, 0, stream>>>(codebook, idx, query_pos, res, curb, qkb);

  // ---- attention block (shared for SA / CA) ----
  const size_t A1 = MDbf;                                // 25,690,112
  const size_t VPADE = (size_t)SP * BATCH * D;           // vp elems incl pad rows
  __bf16* qpb   = (__bf16*)(arena);
  __bf16* kpb   = (__bf16*)(arena + A1);
  __bf16* vpb   = (__bf16*)(arena + 2 * A1);
  float*  scb   = (float*) (arena + 2 * A1 + VPADE * 2); // 16*784*784 f32
  __bf16* attnb = (__bf16*)(arena);                      // reuse (qp dead)
  __bf16* ob    = (__bf16*)(arena + A1);                 // reuse (kp dead)
  float*  aout  = (float*) (arena + 2 * A1 + VPADE * 2); // reuse (sc dead)

  auto attention = [&](const __bf16* qA, const __bf16* kA, const __bf16* vA,
                       const __bf16* wiB, const float* bi,
                       const __bf16* woB, const float* bo) {
    // projections (bf16 out only)
    G(qA, D, 0, wiB,             D, 0, 1, D, bi,         1.0f, 0,
      nullptr, 0, 0, qpb, D, 0, nullptr, 0, M, D, D, D, 1);
    G(kA, D, 0, wiB + (size_t)D * D, D, 0, 1, D, bi + D, 1.0f, 0,
      nullptr, 0, 0, kpb, D, 0, nullptr, 0, M, D, D, D, 1);
    G(vA, D, 0, wiB + (size_t)2 * D * D, D, 0, 1, D, bi + 2 * D, 1.0f, 0,
      nullptr, 0, 0, vpb, D, 0, nullptr, 0, M, D, D, D, 1);
    zero_bf16_k<<<dim3(256), t256, 0, stream>>>(vpb + MD, (long long)(SP - S) * BATCH * D);
    // scores: per batch, qp[s,b,:] . kp[t,b,:] / 32
    G(qpb, (long long)BATCH * D, D, kpb, (long long)BATCH * D, D, 1, S,
      nullptr, 1.0f / 32.0f, 0, scb, S, (long long)S * S,
      nullptr, 0, 0, nullptr, 0, S, S, D, S, BATCH);
    softmax_row<<<dim3(BATCH * S), t256, 0, stream>>>(scb, attnb, S, SP);
    // o = attn @ vp  (B[k,n] layout)
    G(attnb, SP, (long long)S * SP, vpb, (long long)BATCH * D, D, 0, D,
      nullptr, 1.0f, 0, nullptr, 0, 0, ob, (long long)BATCH * D, D,
      nullptr, 0, S, D, SP, D, BATCH);
    // out projection
    G(ob, D, 0, woB, D, 0, 1, D, bo, 1.0f, 0,
      aout, D, 0, nullptr, 0, 0, nullptr, 0, M, D, D, D, 1);
  };

  // self-attention: q=k=tgt+pos, v=tgt ; then LN1 (emit next qkb = tgt1+pos)
  attention(qkb, qkb, curb, sawiB, sa_bi, sawoB, sa_bo);
  ln_res<<<dim3(M), t256, 0, stream>>>(res, aout, n1_g, n1_b, curb, qkb, query_pos, nullptr);

  // cross-attention: q=tgt1+pos, k=v=mem ; then LN2
  attention(qkb, memb, memb, cawiB, ca_bi, cawoB, ca_bo);
  ln_res<<<dim3(M), t256, 0, stream>>>(res, aout, n2_g, n2_b, curb, nullptr, nullptr, nullptr);

  // ---- FFN ----
  {
    __bf16* hB = (__bf16*)(arena);
    float*  ff = (float*)(arena + (size_t)M * HID * 2);
    G(curb, D, 0, l1B, D, 0, 1, HID, lin1_bias, 1.0f, 1,
      nullptr, 0, 0, hB, HID, 0, nullptr, 0, M, HID, D, HID, 1);
    G(hB, HID, 0, l2B, HID, 0, 1, D, lin2_bias, 1.0f, 0,
      ff, D, 0, nullptr, 0, 0, nullptr, 0, M, D, HID, D, 1);
    // tgt3 = mem + LN(tgt2 + ff)
    ln_res<<<dim3(M), t256, 0, stream>>>(res, ff, n3_g, n3_b, curb, nullptr, nullptr, feat);
  }

  // ---- output heads ----
  {
    float*  b1h = (float*) (arena);
    __bf16* h2b = (__bf16*)(arena + (size_t)M * RDP * 4);
    float*  b2h = (float*) (arena + (size_t)M * RDP * 4 + MDbf);
    __bf16* trb = (__bf16*)(arena + (size_t)M * RDP * 4 + MDbf + (size_t)M * RDP * 4);

    G(curb, D, 0, f1B, D, 0, 1, RD, f1_b, 1.0f, 0,
      b1h, RDP, 0, nullptr, 0, 0, nullptr, 0, M, RDP, D, RD, 1);
    G(curb, D, 0, f2aB, D, 0, 1, D, f2a_b, 1.0f, 1,
      nullptr, 0, 0, h2b, D, 0, nullptr, 0, M, D, D, D, 1);
    G(h2b, D, 0, f2bB, D, 0, 1, RD, f2b_b, 1.0f, 0,
      b2h, RDP, 0, nullptr, 0, 0, nullptr, 0, M, RDP, D, RD, 1);
    combine_tr<<<dim3(M), dim3(128), 0, stream>>>(b1h, b2h, out_tr, trb);
    // proj = tr @ ph^T + ph_b  (K padded 90->96, permuted store to (B,P,PD))
    G(trb, RDP, 0, phB, RDP, 0, 1, PD, ph_b, 1.0f, 0,
      nullptr, 0, 0, nullptr, 0, 0, out_proj, PD, M, PD, RDP, PD, 1);
  }
}
